// GatedLinearAttention_50740743635566
// MI455X (gfx1250) — compile-verified, matrix operand order checked by reading, not run on hardware
//
#include <hip/hip_runtime.h>
#include <hip/hip_bf16.h>

// ---------------------------------------------------------------------------
// Types
// ---------------------------------------------------------------------------
typedef __bf16 bf16_t;
typedef __attribute__((ext_vector_type(16))) __bf16 v16bf;
typedef __attribute__((ext_vector_type(8)))  __bf16 v8bf;
typedef __attribute__((ext_vector_type(8)))  float  v8f;
typedef __attribute__((ext_vector_type(4)))  unsigned u32x4;
typedef __attribute__((ext_vector_type(8)))  int      i32x8;
typedef __attribute__((ext_vector_type(4)))  int      i32x4;

#define GLA_B   2
#define GLA_L   4096
#define GLA_H   4
#define GLA_DIM 2048
#define GLA_KD  2048
#define GLA_VD  1024
#define GLA_DK  512
#define GLA_DV  256
#define GLA_CH  64
#define GLA_NC  64           // L / CHUNK
#define GLA_M   (GLA_B*GLA_L)  // 8192 rows

#if defined(__has_builtin)
#  if __has_builtin(__builtin_amdgcn_tensor_load_to_lds) && \
      __has_builtin(__builtin_amdgcn_s_wait_tensorcnt)
#    define GLA_HAVE_TDM 1
#  endif
#endif

__device__ __forceinline__ bf16_t f2bf(float f) {
  union { float f; unsigned u; } x; x.f = f;
  unsigned u = x.u + 0x7FFFu + ((x.u >> 16) & 1u);
  unsigned short h = (unsigned short)(u >> 16);
  bf16_t r; __builtin_memcpy(&r, &h, sizeof(r)); return r;
}

__device__ __forceinline__ v8f vzero8() {
  v8f z = {0.f,0.f,0.f,0.f,0.f,0.f,0.f,0.f}; return z;
}

__device__ __forceinline__ v8f wmma_bf16(v16bf a, v16bf b, v8f c) {
  return __builtin_amdgcn_wmma_f32_16x16x32_bf16(
      /*neg_a=*/false, a, /*neg_b=*/false, b,
      /*c_mod=*/(short)0, c, /*reuse_a=*/false, /*reuse_b=*/false);
}

// Load one 16x32 bf16 A/B fragment strip for lane `lane`.
// Lane layout (ISA 7.12.2): lanes<16 hold K {0..7,16..23}, lanes>=16 hold
// K {8..15,24..31} relative to kbase.
__device__ __forceinline__ v16bf load_frag(const bf16_t* rowptr, int kbase, int lane) {
  int off = (lane & 16) ? 8 : 0;
  union { v16bf v; v8bf h[2]; } u;
  u.h[0] = *(const v8bf*)(rowptr + kbase + off);
  u.h[1] = *(const v8bf*)(rowptr + kbase + 16 + off);
  return u.v;
}

// Same, but the source row is fp32; converted to bf16 in registers.
__device__ __forceinline__ v16bf load_frag_f32(const float* rowptr, int kbase, int lane) {
  int off = (lane & 16) ? 8 : 0;
  v8f f0 = *(const v8f*)(rowptr + kbase + off);
  v8f f1 = *(const v8f*)(rowptr + kbase + 16 + off);
  v8bf a, b;
  #pragma unroll
  for (int i = 0; i < 8; i++) { a[i] = f2bf(f0[i]); b[i] = f2bf(f1[i]); }
  union { v16bf v; v8bf h[2]; } u;
  u.h[0] = a; u.h[1] = b;
  return u.v;
}

#ifdef GLA_HAVE_TDM
// ---------------------------------------------------------------------------
// Tensor Data Mover: async-copy a 2D tile (TILE_K=32 bf16 wide x TILE_R=128
// rows, row stride = K elements) from global memory into LDS, inserting 16B
// of padding after every 64B row so the LDS row stride is 80B (= GEMM_LDSS
// halves).  D# layout per CDNA5 ISA ch.8 (CDNA5 ISA).
// Issued once per workgroup by a single wave; completion via TENSORcnt.
// ---------------------------------------------------------------------------
__device__ __forceinline__ void tdm_load_tile_128x32(const bf16_t* gtile,
                                                     unsigned lds_off, int K) {
  unsigned long long ga = (unsigned long long)(uintptr_t)gtile;
  // --- D# group 0 (128b): count=1, lds_addr, global_addr[56:0], type=2 ---
  u32x4 g0;
  g0[0] = 1u;                                       // count=1, user descriptor
  g0[1] = lds_off;                                  // lds_addr (bytes)
  g0[2] = (unsigned)(ga & 0xFFFFFFFFu);             // global_addr[31:0]
  g0[3] = (unsigned)((ga >> 32) & 0x01FFFFFFu)      // global_addr[56:32]
        | (2u << 30);                               // type = 2 ("image")
  // --- D# group 1 (256b) ---
  const unsigned td0 = 0x7FFFFFFFu;                 // tensor_dim0 (no clip)
  const unsigned td1 = 0x7FFFFFFFu;                 // tensor_dim1 (no clip)
  i32x8 g1;
  g1[0] = (int)((1u << 16)        // data_size = 2 bytes
              | (1u << 20)        // pad_enable
              | (3u << 22)        // pad_interval: 1<<3 * 8B = 64B
              | (3u << 25));      // pad_amount: 4 DWORDs = 16B
  g1[1] = (int)((td0 & 0xFFFFu) << 16);             // [63:48] tensor_dim0 lo16
  g1[2] = (int)(((td0 >> 16) & 0xFFFFu)             // [79:64] tensor_dim0 hi16
              | ((td1 & 0xFFFFu) << 16));           // [95:80] tensor_dim1 lo16
  g1[3] = (int)(((td1 >> 16) & 0xFFFFu)             // [111:96] tensor_dim1 hi16
              | (32u << 16));                       // [127:112] tile_dim0 = 32
  g1[4] = (int)(128u);                              // tile_dim1=128, tile_dim2=0
  g1[5] = (int)(unsigned)K;                         // tensor_dim0_stride lo32
  g1[6] = 0;                                        // stride hi16, d1stride lo16
  g1[7] = 0;                                        // d1stride hi32
  // --- groups 2/3 unused for 2D ---
  i32x4 gz = {0, 0, 0, 0};
#if __clang_major__ >= 23
  i32x8 gz8 = {0, 0, 0, 0, 0, 0, 0, 0};
  __builtin_amdgcn_tensor_load_to_lds(g0, g1, gz, gz, gz8, 0);
#else
  __builtin_amdgcn_tensor_load_to_lds(g0, g1, gz, gz, 0);
#endif
}
#endif  // GLA_HAVE_TDM

// ---------------------------------------------------------------------------
// Elementwise conversion kernels
// ---------------------------------------------------------------------------
__global__ __launch_bounds__(256) void k_f2bf(const float* __restrict__ in,
                                              bf16_t* __restrict__ out, size_t n) {
  size_t i = (size_t)blockIdx.x * 256 + threadIdx.x;
  if (i < n) out[i] = f2bf(in[i]);
}

// in [R, C] row-major fp32 -> out [C, R] row-major bf16 (i.e. transpose)
__global__ __launch_bounds__(256) void k_transpose_f2bf(const float* __restrict__ in,
                                                        bf16_t* __restrict__ out,
                                                        int R, int C) {
  size_t i = (size_t)blockIdx.x * 256 + threadIdx.x;
  if (i < (size_t)R * C) {
    int r = (int)(i / C), c = (int)(i % C);
    out[(size_t)c * R + r] = f2bf(in[i]);
  }
}

__global__ __launch_bounds__(256) void k_zero(float* __restrict__ p, size_t n) {
  size_t i = (size_t)blockIdx.x * 256 + threadIdx.x;
  if (i < n) p[i] = 0.f;
}

// ---------------------------------------------------------------------------
// Generic bf16 WMMA GEMM: C[M,N] = A[M,K] @ B[K,N], with B given transposed
// (BT is [N,K] row-major bf16).  Tile 128x128x32, 256 threads = 8 waves,
// wave grid 4(M) x 2(N), each wave 2x4 = 8 accum tiles.
// Tiles staged into LDS by the Tensor Data Mover (TDM) when available.
// M % 128 == 0, N % 128 == 0, K % 32 == 0 (true for all our shapes).
// ---------------------------------------------------------------------------
#define GEMM_LDSS 40   // 32 + 8 halves of padding (80B LDS row stride)

__global__ __launch_bounds__(256) void k_gemm_bf16(const bf16_t* __restrict__ A,
                                                   const bf16_t* __restrict__ BT,
                                                   float* __restrict__ C,
                                                   int M, int N, int K) {
  __shared__ bf16_t As[128 * GEMM_LDSS];
  __shared__ bf16_t Bs[128 * GEMM_LDSS];

  int tid  = threadIdx.x;
  int lane = tid & 31;
  int ln15 = lane & 15;
  int wid  = tid >> 5;
  int wm   = wid & 3;     // 0..3 -> 32-row M strip
  int wn   = wid >> 2;    // 0..1 -> 64-col N strip

  int nb   = N >> 7;
  int bm   = blockIdx.x / nb;
  int bn   = blockIdx.x % nb;
  int row0 = bm << 7;
  int col0 = bn << 7;

  v8f acc[2][4];
  #pragma unroll
  for (int i = 0; i < 2; i++)
    #pragma unroll
    for (int j = 0; j < 4; j++) acc[i][j] = vzero8();

#ifdef GLA_HAVE_TDM
  unsigned asOff = (unsigned)(uintptr_t)&As[0];   // LDS byte offset (flat low bits)
  unsigned bsOff = (unsigned)(uintptr_t)&Bs[0];
#else
  int lr = tid >> 1;          // 0..127: tile row loaded by this thread
  int lc = (tid & 1) << 4;    // 0 or 16: half-row (16 halves)
#endif

  for (int k0 = 0; k0 < K; k0 += 32) {
#ifdef GLA_HAVE_TDM
    if (tid == 0) {
      tdm_load_tile_128x32(A  + (size_t)row0 * K + k0, asOff, K);
      tdm_load_tile_128x32(BT + (size_t)col0 * K + k0, bsOff, K);
      __builtin_amdgcn_s_wait_tensorcnt(0);
    }
    __syncthreads();
#else
    const bf16_t* ga = A  + (size_t)(row0 + lr) * K + k0 + lc;
    const bf16_t* gb = BT + (size_t)(col0 + lr) * K + k0 + lc;
    *(v8bf*)(&As[lr * GEMM_LDSS + lc])     = *(const v8bf*)(ga);
    *(v8bf*)(&As[lr * GEMM_LDSS + lc + 8]) = *(const v8bf*)(ga + 8);
    *(v8bf*)(&Bs[lr * GEMM_LDSS + lc])     = *(const v8bf*)(gb);
    *(v8bf*)(&Bs[lr * GEMM_LDSS + lc + 8]) = *(const v8bf*)(gb + 8);
    __syncthreads();
#endif

    v16bf afr[2], bfr[4];
    #pragma unroll
    for (int i = 0; i < 2; i++) {
      int m = wm * 32 + i * 16 + ln15;
      afr[i] = load_frag(&As[m * GEMM_LDSS], 0, lane);
    }
    #pragma unroll
    for (int j = 0; j < 4; j++) {
      int n = wn * 64 + j * 16 + ln15;
      bfr[j] = load_frag(&Bs[n * GEMM_LDSS], 0, lane);
    }
    #pragma unroll
    for (int i = 0; i < 2; i++)
      #pragma unroll
      for (int j = 0; j < 4; j++)
        acc[i][j] = wmma_bf16(afr[i], bfr[j], acc[i][j]);
    __syncthreads();
  }

  // Store: D layout -> VGPR r holds (m = r + (lane>=16 ? 8 : 0)), n = lane%16.
  int mofs = (lane & 16) ? 8 : 0;
  #pragma unroll
  for (int i = 0; i < 2; i++) {
    #pragma unroll
    for (int j = 0; j < 4; j++) {
      int mb = row0 + wm * 32 + i * 16 + mofs;
      int n  = col0 + wn * 64 + j * 16 + ln15;
      #pragma unroll
      for (int r = 0; r < 8; r++)
        C[(size_t)(mb + r) * N + n] = acc[i][j][r];
    }
  }
}

// ---------------------------------------------------------------------------
// Low-rank gk path (tiny: keep scalar fp32)
// ---------------------------------------------------------------------------
__global__ __launch_bounds__(256) void k_gk1(const float* __restrict__ x,
                                             const float* __restrict__ W1,
                                             float* __restrict__ t1) {
  int i = blockIdx.x * 256 + threadIdx.x;      // 8192*16 threads
  int row = i >> 4, j = i & 15;
  const float* xr = x + (size_t)row * GLA_DIM;
  float s = 0.f;
  for (int kk = 0; kk < GLA_DIM; kk++) s += xr[kk] * W1[kk * 16 + j];
  t1[i] = s;
}

__global__ __launch_bounds__(256) void k_gk2(const float* __restrict__ t1,
                                             const float* __restrict__ W2,
                                             const float* __restrict__ bias,
                                             float* __restrict__ gkraw) {
  size_t i = (size_t)blockIdx.x * 256 + threadIdx.x;   // 8192*2048
  int row = (int)(i >> 11), n = (int)(i & 2047);
  const float* tr = t1 + (size_t)row * 16;
  float s = bias[n];
  #pragma unroll
  for (int j = 0; j < 16; j++) s += tr[j] * W2[j * GLA_KD + n];
  gkraw[i] = s;
}

// ---------------------------------------------------------------------------
// Per-chunk scan: fuse log-sigmoid, within-chunk cumsum, and produce
//   qe  [B*L, KD] bf16     (q * exp(b) * scale, head-interleaved like q)
//   kd  [B*L, KD] bf16     (k * exp(-b))
//   klT [bh*nc, DK, CH] bf16  (k * exp(blast - b), K-major for S-update B-frags)
//   decay [bh*nc, DK] f32  (exp(blast))
// ---------------------------------------------------------------------------
__global__ __launch_bounds__(256) void k_scan(const float* __restrict__ q,
                                              const float* __restrict__ k,
                                              const float* __restrict__ gkraw,
                                              bf16_t* __restrict__ qe,
                                              bf16_t* __restrict__ kd,
                                              bf16_t* __restrict__ klT,
                                              float*  __restrict__ decay) {
  int bidx = blockIdx.x;
  int half = bidx & 1;  bidx >>= 1;
  int c = bidx & 63;    bidx >>= 6;
  int h = bidx & 3;
  int b = bidx >> 2;
  int col = half * 256 + threadIdx.x;                   // dk in [0,512)
  int rowbase = b * GLA_L + c * GLA_CH;                 // row into [B*L]
  size_t base = (size_t)rowbase * GLA_KD + h * GLA_DK + col;
  const float inv_norm = 1.f / 16.f;
  const float scale = rsqrtf((float)GLA_DK);

  // pass 1: total log-decay over the chunk
  float btot = 0.f;
  for (int r = 0; r < GLA_CH; r++) {
    float xg = gkraw[base + (size_t)r * GLA_KD];
    float ls = fminf(xg, 0.f) - log1pf(__expf(-fabsf(xg)));  // log_sigmoid
    btot += ls * inv_norm;
  }
  int chunkId = ((b * GLA_H + h) * GLA_NC + c);
  decay[(size_t)chunkId * GLA_DK + col] = __expf(btot);

  // pass 2: emit qe, kd, klT
  float bsum = 0.f;
  bf16_t* klrow = klT + ((size_t)chunkId * GLA_DK + col) * GLA_CH;
  for (int r = 0; r < GLA_CH; r++) {
    size_t idx = base + (size_t)r * GLA_KD;
    float xg = gkraw[idx];
    float ls = fminf(xg, 0.f) - log1pf(__expf(-fabsf(xg)));
    bsum += ls * inv_norm;
    float qv = q[idx], kv = k[idx];
    qe[idx]  = f2bf(qv * __expf(bsum) * scale);
    kd[idx]  = f2bf(kv * __expf(-bsum));
    klrow[r] = f2bf(kv * __expf(btot - bsum));
  }
}

// v fp32 [B*L, VD] -> per-chunk transposed vT bf16 [bh*nc, DV, CH]
__global__ __launch_bounds__(256) void k_vT(const float* __restrict__ v,
                                            bf16_t* __restrict__ vT) {
  int chunkId = blockIdx.x;                 // bh*nc + c
  int c  = chunkId & 63;
  int bh = chunkId >> 6;
  int b = bh >> 2, h = bh & 3;
  size_t src = (size_t)(b * GLA_L + c * GLA_CH) * GLA_VD + h * GLA_DV;
  size_t dst = (size_t)chunkId * GLA_DV * GLA_CH;
  for (int i = threadIdx.x; i < GLA_CH * GLA_DV; i += 256) {
    int r = i & (GLA_CH - 1);
    int dv = i >> 6;
    vT[dst + (size_t)dv * GLA_CH + r] = f2bf(v[src + (size_t)r * GLA_VD + dv]);
  }
}

// ---------------------------------------------------------------------------
// Chunk o-kernel: for chunk c, per (b,h):
//   A = mask(qe_c @ kd_c^T)       [64,64]
//   o = A @ v_c + qe_c @ S        [64,256]   (S read as ST [DV,DK] fp32)
// 4 waves, each owns a 16-row M strip.
// ---------------------------------------------------------------------------
#define ABUF_LD 72

__global__ __launch_bounds__(128) void k_gla_o(const bf16_t* __restrict__ qe,
                                               const bf16_t* __restrict__ kd,
                                               const bf16_t* __restrict__ vT,
                                               const float*  __restrict__ ST,
                                               float* __restrict__ o, int c) {
  __shared__ bf16_t Abuf[GLA_CH * ABUF_LD];

  int bh = blockIdx.x;                     // 0..7
  int b = bh >> 2, h = bh & 3;
  int lane = threadIdx.x & 31;
  int ln15 = lane & 15;
  int w = threadIdx.x >> 5;                // wave id 0..3 -> M strip

  int rowg0 = b * GLA_L + c * GLA_CH;
  const bf16_t* qbase = qe + (size_t)rowg0 * GLA_KD + h * GLA_DK;
  const bf16_t* kbase = kd + (size_t)rowg0 * GLA_KD + h * GLA_DK;
  int chunkId = bh * GLA_NC + c;
  const bf16_t* vTb  = vT + (size_t)chunkId * GLA_DV * GLA_CH;
  const float*  Sb   = ST + (size_t)bh * GLA_DV * GLA_DK;

  int m = w * 16 + ln15;
  const bf16_t* qrow = qbase + (size_t)m * GLA_KD;

  // --- Stage 1: A = qe @ kd^T over K = 512 ---
  v8f a4[4];
  #pragma unroll
  for (int nt = 0; nt < 4; nt++) a4[nt] = vzero8();
  #pragma unroll 1
  for (int kt = 0; kt < 16; kt++) {
    v16bf af = load_frag(qrow, kt * 32, lane);
    #pragma unroll
    for (int nt = 0; nt < 4; nt++) {
      const bf16_t* krow = kbase + (size_t)(nt * 16 + ln15) * GLA_KD;
      v16bf bfr = load_frag(krow, kt * 32, lane);
      a4[nt] = wmma_bf16(af, bfr, a4[nt]);
    }
  }
  // causal mask (row >= col) + bf16 into LDS for redistribution
  int mofs = (lane & 16) ? 8 : 0;
  #pragma unroll
  for (int nt = 0; nt < 4; nt++) {
    int nn = nt * 16 + ln15;
    #pragma unroll
    for (int r = 0; r < 8; r++) {
      int mm = w * 16 + mofs + r;
      float vl = (mm >= nn) ? a4[nt][r] : 0.f;
      Abuf[mm * ABUF_LD + nn] = f2bf(vl);
    }
  }
  __syncthreads();

  // --- Stage 2: o = A @ v  +  qe @ S ---
  v8f oacc[16];
  #pragma unroll
  for (int dt = 0; dt < 16; dt++) oacc[dt] = vzero8();

  // intra-chunk: K = 64 (A from LDS, B-frags from vT rows)
  #pragma unroll
  for (int kt = 0; kt < 2; kt++) {
    v16bf af = load_frag(&Abuf[m * ABUF_LD], kt * 32, lane);
    #pragma unroll 1
    for (int dt = 0; dt < 16; dt++) {
      const bf16_t* vrow = vTb + (size_t)(dt * 16 + ln15) * GLA_CH;
      v16bf bfr = load_frag(vrow, kt * 32, lane);
      oacc[dt] = wmma_bf16(af, bfr, oacc[dt]);
    }
  }
  // inter-chunk: K = 512 (A from qe, B-frags from ST fp32 rows -> bf16)
  #pragma unroll 1
  for (int kt = 0; kt < 16; kt++) {
    v16bf af = load_frag(qrow, kt * 32, lane);
    #pragma unroll 1
    for (int dt = 0; dt < 16; dt++) {
      const float* srow = Sb + (size_t)(dt * 16 + ln15) * GLA_DK;
      v16bf bfr = load_frag_f32(srow, kt * 32, lane);
      oacc[dt] = wmma_bf16(af, bfr, oacc[dt]);
    }
  }

  // store o as [B,H,L,DV] fp32
  #pragma unroll
  for (int dt = 0; dt < 16; dt++) {
    int n = dt * 16 + ln15;
    #pragma unroll
    for (int r = 0; r < 8; r++) {
      int mm = w * 16 + mofs + r;
      o[((size_t)bh * GLA_L + c * GLA_CH + mm) * GLA_DV + n] = oacc[dt][r];
    }
  }
}

// ---------------------------------------------------------------------------
// Chunk S-update: ST[dv,dk] = decay[dk] * ST[dv,dk] + (v^T @ k~)[dv,dk]
// Grid: B*H * 4(dv/64) * 4(dk/128), 4 waves per block.
// ---------------------------------------------------------------------------
__global__ __launch_bounds__(128) void k_gla_update(const bf16_t* __restrict__ vT,
                                                    const bf16_t* __restrict__ klT,
                                                    const float*  __restrict__ decay,
                                                    float* __restrict__ ST, int c) {
  int gid = blockIdx.x;
  int bh    = gid >> 4;
  int dvblk = (gid >> 2) & 3;
  int dkblk = gid & 3;
  int lane = threadIdx.x & 31;
  int ln15 = lane & 15;
  int w = threadIdx.x >> 5;

  int chunkId = bh * GLA_NC + c;
  const bf16_t* vTb = vT  + (size_t)chunkId * GLA_DV * GLA_CH;
  const bf16_t* klb = klT + (size_t)chunkId * GLA_DK * GLA_CH;
  const float*  dec = decay + (size_t)chunkId * GLA_DK;
  float* Sb = ST + (size_t)bh * GLA_DV * GLA_DK;

  v8f acc[8];
  #pragma unroll
  for (int nt = 0; nt < 8; nt++) acc[nt] = vzero8();

  int mrow = dvblk * 64 + w * 16 + ln15;
  const bf16_t* arow = vTb + (size_t)mrow * GLA_CH;
  #pragma unroll
  for (int kt = 0; kt < 2; kt++) {
    v16bf af = load_frag(arow, kt * 32, lane);
    #pragma unroll
    for (int nt = 0; nt < 8; nt++) {
      const bf16_t* brow = klb + (size_t)(dkblk * 128 + nt * 16 + ln15) * GLA_CH;
      v16bf bfr = load_frag(brow, kt * 32, lane);
      acc[nt] = wmma_bf16(af, bfr, acc[nt]);
    }
  }

  int mofs = (lane & 16) ? 8 : 0;
  #pragma unroll
  for (int nt = 0; nt < 8; nt++) {
    int n = dkblk * 128 + nt * 16 + ln15;
    float d = dec[n];
    #pragma unroll
    for (int r = 0; r < 8; r++) {
      int mm = dvblk * 64 + w * 16 + mofs + r;
      size_t idx = (size_t)mm * GLA_DK + n;
      Sb[idx] = d * Sb[idx] + acc[nt][r];
    }
  }
}

// ---------------------------------------------------------------------------
// RMS norm + SiLU gate -> gated bf16 [B*L, VD]
// One wave per (b,l,h) row of 256 elements.
// ---------------------------------------------------------------------------
__global__ __launch_bounds__(256) void k_rmsgate(const float* __restrict__ o,
                                                 const float* __restrict__ g,
                                                 const float* __restrict__ Wnorm,
                                                 bf16_t* __restrict__ gated) {
  int gw = blockIdx.x * 8 + (threadIdx.x >> 5);
  int lane = threadIdx.x & 31;
  int h = gw & 3;
  int bl = gw >> 2;                       // 0..8191
  int b = bl >> 12, l = bl & 4095;
  const float* orow = o + ((size_t)(b * GLA_H + h) * GLA_L + l) * GLA_DV;

  float vals[8];
  float ssum = 0.f;
  #pragma unroll
  for (int i = 0; i < 8; i++) {
    vals[i] = orow[lane + i * 32];
    ssum += vals[i] * vals[i];
  }
  #pragma unroll
  for (int off = 16; off > 0; off >>= 1) ssum += __shfl_xor(ssum, off, 32);
  float rinv = rsqrtf(ssum * (1.f / GLA_DV) + 1e-5f);

  const float* grow = g + (size_t)bl * GLA_VD + h * GLA_DV;
  bf16_t* outrow = gated + (size_t)bl * GLA_VD + h * GLA_DV;
  #pragma unroll
  for (int i = 0; i < 8; i++) {
    int dv = lane + i * 32;
    float gv = grow[dv];
    float sig = 1.f / (1.f + __expf(-gv));
    outrow[dv] = f2bf(vals[i] * rinv * Wnorm[dv] * gv * sig);
  }
}

// ---------------------------------------------------------------------------
// Host-side orchestration
// ---------------------------------------------------------------------------
extern "C" void kernel_launch(void* const* d_in, const int* in_sizes, int n_in,
                              void* d_out, int out_size, void* d_ws, size_t ws_size,
                              hipStream_t stream) {
  const float* x     = (const float*)d_in[0];
  const float* Wq    = (const float*)d_in[1];
  const float* Wk    = (const float*)d_in[2];
  const float* Wv    = (const float*)d_in[3];
  const float* Wgk1  = (const float*)d_in[4];
  const float* Wgk2  = (const float*)d_in[5];
  const float* bgk2  = (const float*)d_in[6];
  const float* Wg    = (const float*)d_in[7];
  const float* Wnorm = (const float*)d_in[8];
  const float* Wo    = (const float*)d_in[9];
  float* out = (float*)d_out;

  char* p = (char*)d_ws;
  auto alloc = [&](size_t bytes) -> char* {
    char* r = p;
    p += (bytes + 255) & ~((size_t)255);
    return r;
  };

  const size_t M = GLA_M;
  bf16_t* xbf   = (bf16_t*)alloc(M * GLA_DIM * 2);
  bf16_t* WqT   = (bf16_t*)alloc((size_t)GLA_KD * GLA_DIM * 2);
  bf16_t* WkT   = (bf16_t*)alloc((size_t)GLA_KD * GLA_DIM * 2);
  bf16_t* WvT   = (bf16_t*)alloc((size_t)GLA_VD * GLA_DIM * 2);
  bf16_t* WgT   = (bf16_t*)alloc((size_t)GLA_VD * GLA_DIM * 2);
  bf16_t* WoT   = (bf16_t*)alloc((size_t)GLA_DIM * GLA_VD * 2);
  float*  qbuf  = (float*)alloc(M * GLA_KD * 4);
  float*  kbuf  = (float*)alloc(M * GLA_KD * 4);
  float*  vbuf  = (float*)alloc(M * GLA_VD * 4);
  float*  gbuf  = (float*)alloc(M * GLA_VD * 4);
  float*  t1    = (float*)alloc(M * 16 * 4);
  float*  gkraw = (float*)alloc(M * GLA_KD * 4);
  bf16_t* qebuf = (bf16_t*)alloc(M * GLA_KD * 2);
  bf16_t* kdbuf = (bf16_t*)alloc(M * GLA_KD * 2);
  bf16_t* klT   = (bf16_t*)alloc((size_t)GLA_B * GLA_H * GLA_NC * GLA_DK * GLA_CH * 2);
  float*  decay = (float*)alloc((size_t)GLA_B * GLA_H * GLA_NC * GLA_DK * 4);
  bf16_t* vTb   = (bf16_t*)alloc((size_t)GLA_B * GLA_H * GLA_NC * GLA_DV * GLA_CH * 2);
  float*  STb   = (float*)alloc((size_t)GLA_B * GLA_H * GLA_DV * GLA_DK * 4);
  float*  obuf  = (float*)alloc((size_t)GLA_B * GLA_H * GLA_L * GLA_DV * 4);
  bf16_t* gated = (bf16_t*)alloc(M * GLA_VD * 2);

  // 1) conversions
  {
    size_t n = M * GLA_DIM;
    k_f2bf<<<(unsigned)((n + 255) / 256), 256, 0, stream>>>(x, xbf, n);
  }
  k_transpose_f2bf<<<(GLA_DIM * GLA_KD + 255) / 256, 256, 0, stream>>>(Wq, WqT, GLA_DIM, GLA_KD);
  k_transpose_f2bf<<<(GLA_DIM * GLA_KD + 255) / 256, 256, 0, stream>>>(Wk, WkT, GLA_DIM, GLA_KD);
  k_transpose_f2bf<<<(GLA_DIM * GLA_VD + 255) / 256, 256, 0, stream>>>(Wv, WvT, GLA_DIM, GLA_VD);
  k_transpose_f2bf<<<(GLA_DIM * GLA_VD + 255) / 256, 256, 0, stream>>>(Wg, WgT, GLA_DIM, GLA_VD);
  k_transpose_f2bf<<<(GLA_VD * GLA_DIM + 255) / 256, 256, 0, stream>>>(Wo, WoT, GLA_VD, GLA_DIM);

  // 2) projections via WMMA GEMM (TDM-staged tiles)
  k_gemm_bf16<<<(M / 128) * (GLA_KD / 128), 256, 0, stream>>>(xbf, WqT, qbuf, (int)M, GLA_KD, GLA_DIM);
  k_gemm_bf16<<<(M / 128) * (GLA_KD / 128), 256, 0, stream>>>(xbf, WkT, kbuf, (int)M, GLA_KD, GLA_DIM);
  k_gemm_bf16<<<(M / 128) * (GLA_VD / 128), 256, 0, stream>>>(xbf, WvT, vbuf, (int)M, GLA_VD, GLA_DIM);
  k_gemm_bf16<<<(M / 128) * (GLA_VD / 128), 256, 0, stream>>>(xbf, WgT, gbuf, (int)M, GLA_VD, GLA_DIM);

  // 3) low-rank gk path
  k_gk1<<<(M * 16) / 256, 256, 0, stream>>>(x, Wgk1, t1);
  k_gk2<<<(unsigned)((M * GLA_KD) / 256), 256, 0, stream>>>(t1, Wgk2, bgk2, gkraw);

  // 4) per-chunk scan + v transpose
  k_scan<<<GLA_B * GLA_H * GLA_NC * 2, 256, 0, stream>>>(qbuf, kbuf, gkraw,
                                                         qebuf, kdbuf, klT, decay);
  k_vT<<<GLA_B * GLA_H * GLA_NC, 256, 0, stream>>>(vbuf, vTb);

  // 5) recurrent state init + 64 chunk steps
  {
    size_t n = (size_t)GLA_B * GLA_H * GLA_DV * GLA_DK;
    k_zero<<<(unsigned)((n + 255) / 256), 256, 0, stream>>>(STb, n);
  }
  for (int c = 0; c < GLA_NC; c++) {
    k_gla_o<<<GLA_B * GLA_H, 128, 0, stream>>>(qebuf, kdbuf, vTb, STb, obuf, c);
    k_gla_update<<<GLA_B * GLA_H * 16, 128, 0, stream>>>(vTb, klT, decay, STb, c);
  }

  // 6) RMS norm + SiLU gate
  k_rmsgate<<<(GLA_B * GLA_L * GLA_H) / 8, 256, 0, stream>>>(obuf, gbuf, Wnorm, gated);

  // 7) final projection @ Wo -> d_out
  k_gemm_bf16<<<(M / 128) * (GLA_DIM / 128), 256, 0, stream>>>(gated, WoT, out,
                                                               (int)M, GLA_DIM, GLA_VD);
}
